// LSTMModel_3616362463479
// MI455X (gfx1250) — matrix operand order repaired and checked by reading, not compiled, not run on hardware
//
#include <hip/hip_runtime.h>

typedef _Float16 v16h __attribute__((ext_vector_type(16)));
typedef _Float16 v8h  __attribute__((ext_vector_type(8)));
typedef float    v8f  __attribute__((ext_vector_type(8)));
typedef unsigned int v4u __attribute__((ext_vector_type(4)));
typedef int      v8i  __attribute__((ext_vector_type(8)));
typedef int      v4i  __attribute__((ext_vector_type(4)));

#define B_TOT 2048
#define SEQ   512
#define PRED  96

// ---- workspace / LDS layout -------------------------------------------------
// 140 weight fragments, each 512 halves, [frag][lane(32)][half(16)] in the
// exact wave32 WMMA B layout:
//   f   0..31 : Whh0   (16 n-tiles x 2 k-tiles)
//   f  32..63 : Wih1
//   f  64..95 : Whh1
//   f  96..107: fcW    (6 n-tiles x 2 k-tiles)
//   f 108..123: AFF0   (16 n-tiles, k-tile 0 only; row0=Wih0, row1=bih0+bhh0)
//   f 124..139: AFF1   (16 n-tiles, k-tile 0 only; row1=bih1+bhh1)
#define WS_FRAG_HALVES (140*512)             // 71680
#define WS_FLOATS      96                    // fcb
#define WS_FLOAT_OFF_B (WS_FRAG_HALVES*2)    // 143360 bytes
#define LDS_MIRROR_B   (WS_FLOAT_OFF_B + WS_FLOATS*4)  // 143744 bytes (mult of 8)
#define WAVES_PER_WG   2
#define LDS_TOTAL_B    (LDS_MIRROR_B + WAVES_PER_WG*4096)

#define FR_WHH0 0
#define FR_WIH1 16384
#define FR_WHH1 32768
#define FR_FCW  49152
#define FR_AFF0 55296
#define FR_AFF1 63488

// ---- prep: swizzle f32 weights into f16 WMMA B-fragments in ws --------------
__global__ void lstm_prep(const float* __restrict__ Wih0,
                          const float* __restrict__ Whh0,
                          const float* __restrict__ bih0,
                          const float* __restrict__ bhh0,
                          const float* __restrict__ Wih1,
                          const float* __restrict__ Whh1,
                          const float* __restrict__ bih1,
                          const float* __restrict__ bhh1,
                          const float* __restrict__ fcW,
                          const float* __restrict__ fcb,
                          _Float16* __restrict__ wsH,
                          float* __restrict__ wsF)
{
    int idx = blockIdx.x * 256 + threadIdx.x;
    if (idx < WS_FRAG_HALVES) {
        int f    = idx >> 9;        // global fragment id 0..139
        int r    = idx & 511;
        int lane = r >> 4;
        int h    = r & 15;
        // B element (K=k, N=g): lanes 0-15 hold K=0..7,16..23 of column N=lane,
        // lanes 16-31 hold K=8..15,24..31 of column N=lane-16.
        int col  = lane & 15;
        int krow = (h < 8) ? (((lane >> 4) << 3) + h) : (16 + ((lane >> 4) << 3) + (h - 8));   // K within a 32-row k-tile
        float v;
        if (f < 108) {
            const float* W; int lf;
            if      (f < 32) { W = Whh0; lf = f;      }
            else if (f < 64) { W = Wih1; lf = f - 32; }
            else if (f < 96) { W = Whh1; lf = f - 64; }
            else             { W = fcW;  lf = f - 96; }
            int nt = lf >> 1, kt = lf & 1;
            int g  = nt * 16 + col;
            v = W[g * 64 + kt * 32 + krow];  // all these matrices are (G,64)
        } else if (f < 124) {                // AFF0: row0 = Wih0, row1 = bias0
            int g = (f - 108) * 16 + col;
            v = (krow == 0) ? Wih0[g] : (krow == 1 ? (bih0[g] + bhh0[g]) : 0.0f);
        } else {                             // AFF1: row1 = bias1
            int g = (f - 124) * 16 + col;
            v = (krow == 1) ? (bih1[g] + bhh1[g]) : 0.0f;
        }
        wsH[idx] = (_Float16)v;
    } else if (idx < WS_FRAG_HALVES + WS_FLOATS) {
        wsF[idx - WS_FRAG_HALVES] = fcb[idx - WS_FRAG_HALVES];
    }
}

// ---- fast gate math ---------------------------------------------------------
__device__ __forceinline__ float fexp2(float x) { return __builtin_amdgcn_exp2f(x); }
__device__ __forceinline__ float frcp(float x)  { return __builtin_amdgcn_rcpf(x); }
__device__ __forceinline__ float fsig(float x)  { return frcp(1.0f + fexp2(x * -1.44269504f)); }
__device__ __forceinline__ float ftanhf_(float x){ return 1.0f - 2.0f * frcp(fexp2(x * 2.88539008f) + 1.0f); }

__device__ __forceinline__ v16h combine16(v8h lo, v8h hi) {
    return __builtin_shufflevector(lo, hi, 0,1,2,3,4,5,6,7,8,9,10,11,12,13,14,15);
}

// ---- main fused 2-layer LSTM + FC ------------------------------------------
__global__ __launch_bounds__(WAVES_PER_WG * 32)
void lstm_main(const float* __restrict__ x,
               const void*  __restrict__ ws,
               float* __restrict__ out)
{
    extern __shared__ char smem[];
    _Float16* sFrag = (_Float16*)smem;
    float*    sFcb  = (float*)(smem + WS_FLOAT_OFF_B);

    const int tid = threadIdx.x;

    // ---- TDM: DMA the whole weight mirror (143744 B) global -> LDS ----------
    // 1-D tensor, 8-byte elements, one row of 17968 qwords. Wave 0 issues the
    // descriptor (TDM ignores EXEC; branch is wave-uniform), waits TENSORcnt,
    // then the workgroup barrier releases everyone.
    if (tid < 32) {
        const unsigned long long ga = (unsigned long long)ws;
        const unsigned n8 = LDS_MIRROR_B / 8;          // 17968 (< 65536)
        v4u g0;
        g0[0] = 1u;                                    // count=1 (valid), load
        g0[1] = 0u;                                    // lds_addr = 0 (dyn LDS base)
        g0[2] = (unsigned)(ga & 0xFFFFFFFFu);          // global_addr[31:0]
        g0[3] = (unsigned)((ga >> 32) & 0x01FFFFFFu)   // global_addr[56:32]
              | (2u << 30);                            // type = 2 ("image")
        v8i g1;
        g1[0] = (3 << 16);                             // mask=0, data_size=3 (8B)
        g1[1] = (int)((n8 & 0xFFFFu) << 16);           // tensor_dim0[15:0]
        g1[2] = (int)((n8 >> 16) | (1u << 16));        // tensor_dim0[31:16] | tensor_dim1=1
        g1[3] = (int)((n8 & 0xFFFFu) << 16);           // tile_dim0 = n8
        g1[4] = 1;                                     // tile_dim1 = 1, tile_dim2 = 0
        g1[5] = (int)n8;                               // tensor_dim0_stride[31:0]
        g1[6] = 0;                                     // stride0[47:32] | stride1[15:0]
        g1[7] = 0;
        v4i zz4; zz4[0] = 0; zz4[1] = 0; zz4[2] = 0; zz4[3] = 0;
        v8i zz8;
#pragma unroll
        for (int i = 0; i < 8; ++i) zz8[i] = 0;
        __builtin_amdgcn_tensor_load_to_lds(g0, g1, zz4, zz4, zz8, 0);
        __builtin_amdgcn_s_wait_tensorcnt(0);
    }
    __syncthreads();

    const int wave   = tid >> 5;
    const int lane   = tid & 31;
    const int colIdx = lane & 15;          // N column / A row / D column
    const int rhi    = (lane >> 4) << 3;   // +8 for upper half-wave (rows & A K-chunk)
    const int b0     = blockIdx.x * (WAVES_PER_WG * 16) + wave * 16;

    _Float16* bufA = (_Float16*)(smem + LDS_MIRROR_B) + wave * 2048;  // h of layer0, [16][64]
    _Float16* bufB = bufA + 1024;                                     // h of layer1, [16][64]

    v16h aA[2], aB[2];              // h as WMMA A fragments (16x64 f16 -> 2 frags)
    v16h aX;                        // affine A frag: K0 = x[b,t], K1 = 1.0 (lanes 0-15)
    float c0[4][8], c1[4][8];
#pragma unroll
    for (int h = 0; h < 16; ++h) {
        aA[0][h] = (_Float16)0.0f; aA[1][h] = (_Float16)0.0f;
        aB[0][h] = (_Float16)0.0f; aB[1][h] = (_Float16)0.0f;
        aX[h]    = (_Float16)0.0f;
    }
    aX[1] = (lane < 16) ? (_Float16)1.0f : (_Float16)0.0f;
#pragma unroll
    for (int j = 0; j < 4; ++j)
#pragma unroll
        for (int m = 0; m < 8; ++m) { c0[j][m] = 0.0f; c1[j][m] = 0.0f; }

    v8f zero8;
#pragma unroll
    for (int m = 0; m < 8; ++m) zero8[m] = 0.0f;

    const float* xrow = x + (b0 + colIdx) * SEQ;   // one x scalar per lane per step

#pragma unroll 1
    for (int t = 0; t < SEQ; ++t) {
        const float xv = xrow[t];
        aX[0] = (lane < 16) ? (_Float16)xv : (_Float16)0.0f;

        // ================= layer 0 =================
#pragma unroll
        for (int jt = 0; jt < 4; ++jt) {
            v8f acc[4];
#pragma unroll
            for (int gt = 0; gt < 4; ++gt) {
                const int nt = gt * 4 + jt;
                {   // acc = x*Wih0 + bias  via affine fragment
                    const v16h bf = *(const v16h*)(sFrag + FR_AFF0 + nt * 512 + lane * 16);
                    acc[gt] = __builtin_amdgcn_wmma_f32_16x16x32_f16(false, aX, false, bf,
                                                                     (short)0, zero8, false, false);
                }
#pragma unroll
                for (int kt = 0; kt < 2; ++kt) {   // + h0_{t-1} @ Whh0^T
                    const v16h bf = *(const v16h*)(sFrag + FR_WHH0 + (nt * 2 + kt) * 512 + lane * 16);
                    acc[gt] = __builtin_amdgcn_wmma_f32_16x16x32_f16(false, aA[kt], false, bf,
                                                                     (short)0, acc[gt], false, false);
                }
            }
#pragma unroll
            for (int m = 0; m < 8; ++m) {
                const float ig = fsig(acc[0][m]);
                const float fg = fsig(acc[1][m]);
                const float gg = ftanhf_(acc[2][m]);
                const float og = fsig(acc[3][m]);
                const float c  = fg * c0[jt][m] + ig * gg;
                c0[jt][m] = c;
                bufA[(rhi + m) * 64 + jt * 16 + colIdx] = (_Float16)(og * ftanhf_(c));
            }
        }
        // reload layer-0 h as A fragments (D layout -> A layout via LDS)
#pragma unroll
        for (int kt = 0; kt < 2; ++kt) {
            const _Float16* p = bufA + colIdx * 64 + kt * 32 + rhi;
            aA[kt] = combine16(*(const v8h*)p, *(const v8h*)(p + 16));
        }

        // ================= layer 1 =================
#pragma unroll
        for (int jt = 0; jt < 4; ++jt) {
            v8f acc[4];
#pragma unroll
            for (int gt = 0; gt < 4; ++gt) {
                const int nt = gt * 4 + jt;
                {   // acc = bias1 via affine fragment (x row is zero in B)
                    const v16h bf = *(const v16h*)(sFrag + FR_AFF1 + nt * 512 + lane * 16);
                    acc[gt] = __builtin_amdgcn_wmma_f32_16x16x32_f16(false, aX, false, bf,
                                                                     (short)0, zero8, false, false);
                }
#pragma unroll
                for (int kt = 0; kt < 2; ++kt) {   // + h0_t @ Wih1^T
                    const v16h bf = *(const v16h*)(sFrag + FR_WIH1 + (nt * 2 + kt) * 512 + lane * 16);
                    acc[gt] = __builtin_amdgcn_wmma_f32_16x16x32_f16(false, aA[kt], false, bf,
                                                                     (short)0, acc[gt], false, false);
                }
#pragma unroll
                for (int kt = 0; kt < 2; ++kt) {   // + h1_{t-1} @ Whh1^T
                    const v16h bf = *(const v16h*)(sFrag + FR_WHH1 + (nt * 2 + kt) * 512 + lane * 16);
                    acc[gt] = __builtin_amdgcn_wmma_f32_16x16x32_f16(false, aB[kt], false, bf,
                                                                     (short)0, acc[gt], false, false);
                }
            }
#pragma unroll
            for (int m = 0; m < 8; ++m) {
                const float ig = fsig(acc[0][m]);
                const float fg = fsig(acc[1][m]);
                const float gg = ftanhf_(acc[2][m]);
                const float og = fsig(acc[3][m]);
                const float c  = fg * c1[jt][m] + ig * gg;
                c1[jt][m] = c;
                bufB[(rhi + m) * 64 + jt * 16 + colIdx] = (_Float16)(og * ftanhf_(c));
            }
        }
#pragma unroll
        for (int kt = 0; kt < 2; ++kt) {
            const _Float16* p = bufB + colIdx * 64 + kt * 32 + rhi;
            aB[kt] = combine16(*(const v8h*)p, *(const v8h*)(p + 16));
        }

        // ================= FC on last step =================
        if (t == SEQ - 1) {
#pragma unroll
            for (int pt = 0; pt < 6; ++pt) {
                v8f acc;
                const float bb = sFcb[pt * 16 + colIdx];
#pragma unroll
                for (int m = 0; m < 8; ++m) acc[m] = bb;
#pragma unroll
                for (int kt = 0; kt < 2; ++kt) {
                    const v16h bf = *(const v16h*)(sFrag + FR_FCW + (pt * 2 + kt) * 512 + lane * 16);
                    acc = __builtin_amdgcn_wmma_f32_16x16x32_f16(false, aB[kt], false, bf,
                                                                 (short)0, acc, false, false);
                }
#pragma unroll
                for (int m = 0; m < 8; ++m)
                    out[(b0 + rhi + m) * PRED + pt * 16 + colIdx] = acc[m];
            }
        }
    }
}

extern "C" void kernel_launch(void* const* d_in, const int* in_sizes, int n_in,
                              void* d_out, int out_size, void* d_ws, size_t ws_size,
                              hipStream_t stream)
{
    (void)in_sizes; (void)n_in; (void)out_size; (void)ws_size;

    const float* x    = (const float*)d_in[0];
    const float* Wih0 = (const float*)d_in[1];
    const float* Whh0 = (const float*)d_in[2];
    const float* bih0 = (const float*)d_in[3];
    const float* bhh0 = (const float*)d_in[4];
    const float* Wih1 = (const float*)d_in[5];
    const float* Whh1 = (const float*)d_in[6];
    const float* bih1 = (const float*)d_in[7];
    const float* bhh1 = (const float*)d_in[8];
    const float* fcW  = (const float*)d_in[9];
    const float* fcb  = (const float*)d_in[10];

    _Float16* wsH = (_Float16*)d_ws;
    float*    wsF = (float*)((char*)d_ws + WS_FLOAT_OFF_B);

    const int prepThreads = WS_FRAG_HALVES + WS_FLOATS;
    lstm_prep<<<(prepThreads + 255) / 256, 256, 0, stream>>>(
        Wih0, Whh0, bih0, bhh0, Wih1, Whh1, bih1, bhh1, fcW, fcb, wsH, wsF);

    lstm_main<<<B_TOT / (WAVES_PER_WG * 16), WAVES_PER_WG * 32, LDS_TOTAL_B, stream>>>(
        x, d_ws, (float*)d_out);
}
